// SophisticatedBioInspiredModel_24730421690970
// MI455X (gfx1250) — compile-verified, matrix operand order checked by reading, not run on hardware
//
#include <hip/hip_runtime.h>
#include <hip/hip_bf16.h>
#include <math.h>

// ---------------------------------------------------------------------------
// SophisticatedBioInspiredModel for MI455X (gfx1250, wave32, WMMA, async-LDS)
//
// All heavy GEMMs run in f16 operands / f32 accumulate via
// v_wmma_f32_16x16x32_f16.  Weights are transposed once per call into
// [N][K] f16 so both A and B tiles stage into LDS as pure row-major 16-byte
// copies -- which lets the tile staging use GLOBAL_LOAD_ASYNC_TO_LDS_B128
// (ASYNCcnt-tracked DMA, no VGPR round trip) with double buffering.
// ---------------------------------------------------------------------------

typedef __attribute__((ext_vector_type(16))) _Float16 v16h;
typedef __attribute__((ext_vector_type(8)))  _Float16 v8h;
typedef __attribute__((ext_vector_type(8)))  float    v8f;
typedef __attribute__((ext_vector_type(4)))  int      v4i;

#define B_ROWS   8192
#define D_IN_    1024
#define HID_     2048
#define ENH_     2176
#define N_EXP_   16
#define EXP_DIM_ 128
#define N_CLS_   1000
#define VOCAB_   32000

#if __has_builtin(__builtin_amdgcn_global_load_async_to_lds_b128) && \
    __has_builtin(__builtin_amdgcn_s_wait_asynccnt)
#define USE_ASYNC_LDS 1
#else
#define USE_ASYNC_LDS 0
#endif

__device__ __forceinline__ void stage16(_Float16* lds_dst, const _Float16* g_src)
{
#if USE_ASYNC_LDS
    __builtin_amdgcn_global_load_async_to_lds_b128(
        (__attribute__((address_space(1))) v4i*)(g_src),
        (__attribute__((address_space(3))) v4i*)(lds_dst),
        0, 0);
#else
    *reinterpret_cast<v8h*>(lds_dst) = *reinterpret_cast<const v8h*>(g_src);
#endif
}

__device__ __forceinline__ void stage_wait()
{
#if USE_ASYNC_LDS
    __builtin_amdgcn_s_wait_asynccnt(0);
#endif
}

// ---------------------------------------------------------------------------
// Tiled WMMA GEMM:  C[M,N] = act(A[M,K] (f16) * BT[N,K]^T (f16) + bias[N])
//   block = 256 threads = 8 waves; tile 128x128, K-step 32, double-buffered
//   LDS.  Wave (wm 0..3, wn 0..1) owns rows wm*32..+32, cols wn*64..+64
//   => 2x4 grid of v_wmma_f32_16x16x32_f16, fp32 accumulate.
//   A is [M][K] row-major (lda=K); BT is [N][K] row-major (ld=K), so both
//   tiles stage identically: 128 rows x 32 halfs (4 x 16B chunks per row).
// ---------------------------------------------------------------------------
__global__ __launch_bounds__(256) void wmma_gemm_f16(
    const _Float16* __restrict__ A, const _Float16* __restrict__ BT,
    const float* __restrict__ bias, void* __restrict__ Cout,
    int M, int N, int K, int ldc, int do_relu, int c_is_f16)
{
    const int tid  = threadIdx.x;
    const int lane = tid & 31;
    const int w    = tid >> 5;      // 0..7
    const int wm   = w >> 1;        // 0..3  (M direction)
    const int wn   = w & 1;         // 0..1  (N direction)
    const int m0   = blockIdx.y * 128;
    const int n0   = blockIdx.x * 128;

    __shared__ alignas(16) _Float16 sA[2][128][32];   // [m][k]
    __shared__ alignas(16) _Float16 sB[2][128][32];   // [n][k]

    v8f acc[2][4] = {};

    const int rsel = lane & 15;
    const int khi  = (lane >> 4) & 1;

    // each thread stages 2 x 16B chunks of A and 2 x 16B of B per k-step
    auto stage_tiles = [&](int buf, int k0) {
        #pragma unroll
        for (int p = 0; p < 2; ++p) {
            const int id  = tid + (p << 8);        // 0..511
            const int row = id >> 2;               // 0..127
            const int co  = (id & 3) * 8;          // 0,8,16,24 halfs
            stage16(&sA[buf][row][co], A + (size_t)(m0 + row) * K + k0 + co);
            const int nr = n0 + row;
            if (nr < N)                            // guard (N=1000 tail tile)
                stage16(&sB[buf][row][co], BT + (size_t)nr * K + k0 + co);
        }
    };

    stage_tiles(0, 0);

    const int nIter = K / 32;
    for (int it = 0; it < nIter; ++it) {
        const int cur = it & 1;
        stage_wait();                 // async copies for `cur` complete
        __syncthreads();              // ..visible to all waves; prev reads done
        if (it + 1 < nIter)
            stage_tiles(cur ^ 1, (it + 1) * 32);  // prefetch next tile

        // fragments (ISA 16-bit A layout: lane = M%16, lane[4] selects the
        // +8 K-half; VGPRs 0-3 / 4-7 cover K 0-15 / 16-31)
        v16h afrag[2], bfrag[4];
        #pragma unroll
        for (int mi = 0; mi < 2; ++mi) {
            const int row = wm * 32 + mi * 16 + rsel;
            const v8h lo = *reinterpret_cast<const v8h*>(&sA[cur][row][khi * 8]);
            const v8h hi = *reinterpret_cast<const v8h*>(&sA[cur][row][16 + khi * 8]);
            afrag[mi] = __builtin_shufflevector(lo, hi,
                0, 1, 2, 3, 4, 5, 6, 7, 8, 9, 10, 11, 12, 13, 14, 15);
        }
        #pragma unroll
        for (int ni = 0; ni < 4; ++ni) {
            const int col = wn * 64 + ni * 16 + rsel;
            const v8h lo = *reinterpret_cast<const v8h*>(&sB[cur][col][khi * 8]);
            const v8h hi = *reinterpret_cast<const v8h*>(&sB[cur][col][16 + khi * 8]);
            bfrag[ni] = __builtin_shufflevector(lo, hi,
                0, 1, 2, 3, 4, 5, 6, 7, 8, 9, 10, 11, 12, 13, 14, 15);
        }
        #pragma unroll
        for (int mi = 0; mi < 2; ++mi)
            #pragma unroll
            for (int ni = 0; ni < 4; ++ni)
                acc[mi][ni] = __builtin_amdgcn_wmma_f32_16x16x32_f16(
                    false, afrag[mi], false, bfrag[ni],
                    (short)0, acc[mi][ni], false, false);
    }

    // epilogue: bias + optional relu; C/D layout: lane = N%16, VGPR v ->
    // M = v + 8*lane[4]
    float*    Cf = (float*)Cout;
    _Float16* Ch = (_Float16*)Cout;
    #pragma unroll
    for (int mi = 0; mi < 2; ++mi) {
        #pragma unroll
        for (int ni = 0; ni < 4; ++ni) {
            const int col = n0 + wn * 64 + ni * 16 + rsel;
            const float bv = (col < N) ? bias[col] : 0.f;
            #pragma unroll
            for (int v = 0; v < 8; ++v) {
                const int row = m0 + wm * 32 + mi * 16 + v + khi * 8;
                float x = acc[mi][ni][v] + bv;
                if (do_relu) x = fmaxf(x, 0.f);
                if (col < N) {
                    if (c_is_f16) Ch[(size_t)row * ldc + col] = (_Float16)x;
                    else          Cf[(size_t)row * ldc + col] = x;
                }
            }
        }
    }
}

// ---------------------------------------------------------------------------
// f32 -> f16 convert (for x)
// ---------------------------------------------------------------------------
__global__ __launch_bounds__(256) void convert_f16_kernel(
    const float* __restrict__ in, _Float16* __restrict__ out)
{
    const int i = blockIdx.x * 256 + threadIdx.x;
    out[i] = (_Float16)in[i];
}

// ---------------------------------------------------------------------------
// f32 [K][N] -> f16 [N][K] transpose (weights)
// ---------------------------------------------------------------------------
__global__ __launch_bounds__(256) void transpose_f16_kernel(
    const float* __restrict__ in, _Float16* __restrict__ out, int K, int N)
{
    const int i = blockIdx.x * 256 + threadIdx.x;   // over N*K
    const int n = i / K;
    const int k = i - n * K;
    out[i] = (_Float16)in[(size_t)k * N + n];
}

// ---------------------------------------------------------------------------
// Phasor bank: per-row mean of enh[:, :2048], then 64 sin + 64 cos harmonics
// ---------------------------------------------------------------------------
__global__ __launch_bounds__(256) void phasor_kernel(_Float16* __restrict__ enh)
{
    const int b   = blockIdx.x;
    const int tid = threadIdx.x;
    __shared__ float red[256];
    const _Float16* p = enh + (size_t)b * ENH_;
    float s = 0.f;
    for (int j = tid; j < HID_; j += 256) s += (float)p[j];
    red[tid] = s;
    __syncthreads();
    for (int st = 128; st > 0; st >>= 1) {
        if (tid < st) red[tid] += red[tid + st];
        __syncthreads();
    }
    const float mean = red[0] * (1.0f / (float)HID_);
    if (tid < 64) {
        const float ph = 7.0f * mean * (float)(tid + 1);
        enh[(size_t)b * ENH_ + HID_ + tid] = (_Float16)sinf(ph);
    } else if (tid < 128) {
        const int h = tid - 64;
        const float ph = 7.0f * mean * (float)(h + 1);
        enh[(size_t)b * ENH_ + HID_ + 64 + h] = (_Float16)cosf(ph);
    }
}

// ---------------------------------------------------------------------------
// Gate: one wave per row, 16 logits, wave-wide __shfl_xor reduce, softmax
// ---------------------------------------------------------------------------
__global__ __launch_bounds__(256) void gate_softmax_kernel(
    const _Float16* __restrict__ enh, const float* __restrict__ Wg,
    const float* __restrict__ bg, float* __restrict__ gate)
{
    const int row  = (blockIdx.x * 256 + threadIdx.x) >> 5;
    const int lane = threadIdx.x & 31;
    if (row >= B_ROWS) return;
    const _Float16* a = enh + (size_t)row * ENH_;
    float acc[N_EXP_];
    #pragma unroll
    for (int e = 0; e < N_EXP_; ++e) acc[e] = 0.f;
    for (int k = lane; k < ENH_; k += 32) {
        const float av = (float)a[k];
        const float* wr = Wg + (size_t)k * N_EXP_;
        #pragma unroll
        for (int e = 0; e < N_EXP_; ++e) acc[e] += av * wr[e];
    }
    #pragma unroll
    for (int e = 0; e < N_EXP_; ++e)
        for (int off = 16; off > 0; off >>= 1)
            acc[e] += __shfl_xor(acc[e], off, 32);
    float m = -INFINITY;
    #pragma unroll
    for (int e = 0; e < N_EXP_; ++e) { acc[e] += bg[e]; m = fmaxf(m, acc[e]); }
    float s = 0.f;
    #pragma unroll
    for (int e = 0; e < N_EXP_; ++e) { acc[e] = expf(acc[e] - m); s += acc[e]; }
    if (lane < N_EXP_)
        gate[(size_t)row * N_EXP_ + lane] = acc[lane] / s;
}

// ---------------------------------------------------------------------------
// combined[b,k] = sum_e gate[b,e] * expert[b, e*128 + k]
// ---------------------------------------------------------------------------
__global__ __launch_bounds__(256) void moe_combine_kernel(
    const float* __restrict__ gate, const _Float16* __restrict__ expert,
    _Float16* __restrict__ comb)
{
    const int i  = blockIdx.x * 256 + threadIdx.x;   // < B*128
    const int b  = i >> 7;
    const int kk = i & 127;
    const float* g = gate + (size_t)b * N_EXP_;
    const _Float16* ex = expert + (size_t)b * HID_ + kk;
    float s = 0.f;
    #pragma unroll
    for (int e = 0; e < N_EXP_; ++e) s += g[e] * (float)ex[e * EXP_DIM_];
    comb[i] = (_Float16)s;
}

// ---------------------------------------------------------------------------
// Spiking top-k on context[0] (only the incremented token can cross theta
// each step; top_k ties -> lower index; count-0 fill from lowest vocab
// indices).  Single block; serial tail on thread 0.
// ---------------------------------------------------------------------------
__global__ __launch_bounds__(256) void spike_topk_kernel(
    const _Float16* __restrict__ ctx, float* __restrict__ gains)
{
    __shared__ float vals[HID_];
    __shared__ float sv[256];
    __shared__ int   si[256];
    __shared__ int   tokd[100];
    const int tid = threadIdx.x;

    for (int j = tid; j < HID_; j += 256) {
        vals[j]  = fabsf((float)ctx[j]);
        gains[j] = 1.0f;
    }
    __syncthreads();

    for (int r = 0; r < 100; ++r) {
        float bv = -1.f; int bi = -1;
        for (int j = tid; j < HID_; j += 256) {
            const float v = vals[j];
            if (v > bv || (v == bv && j > bi)) { bv = v; bi = j; }
        }
        sv[tid] = bv; si[tid] = bi;
        __syncthreads();
        for (int st = 128; st > 0; st >>= 1) {
            if (tid < st) {
                const float ov = sv[tid + st]; const int oi = si[tid + st];
                if (ov > sv[tid] || (ov == sv[tid] && oi > si[tid])) {
                    sv[tid] = ov; si[tid] = oi;
                }
            }
            __syncthreads();
        }
        if (tid == 0) { tokd[r] = si[0]; vals[si[0]] = -2.f; }
        __syncthreads();
    }

    if (tid == 0) {
        int utok[100]; float uv[100]; int ucnt[100]; int u = 0;
        for (int i = 99; i >= 0; --i) {            // ascending-|value| order
            const int ti = tokd[i] % VOCAB_;
            for (int j = 0; j < u; ++j) uv[j] *= 0.8f;
            int slot = -1;
            for (int j = 0; j < u; ++j)
                if (utok[j] == ti) { slot = j; break; }
            if (slot < 0) { slot = u; utok[u] = ti; uv[u] = 0.f; ucnt[u] = 0; ++u; }
            uv[slot] += 1.0f;
            if (uv[slot] > 1.2f) { ucnt[slot]++; uv[slot] = 0.f; }
        }
        int picked[20]; int np = 0; int z = 0;
        for (int p = 0; p < 20; ++p) {
            int bestj = -1;
            for (int j = 0; j < u; ++j) {
                if (ucnt[j] <= 0) continue;
                bool used = false;
                for (int q = 0; q < np; ++q)
                    if (picked[q] == utok[j]) { used = true; break; }
                if (used) continue;
                if (bestj < 0 || ucnt[j] > ucnt[bestj] ||
                    (ucnt[j] == ucnt[bestj] && utok[j] < utok[bestj]))
                    bestj = j;
            }
            int sel;
            if (bestj >= 0) {
                sel = utok[bestj];
            } else {
                for (;;) {
                    bool used = false;
                    for (int q = 0; q < np; ++q)
                        if (picked[q] == z) { used = true; break; }
                    if (!used) break;
                    ++z;
                }
                sel = z++;
            }
            picked[np++] = sel;
            if (sel < HID_) gains[sel] += 1.0f;
        }
    }
}

// ---------------------------------------------------------------------------
// context[b, j] *= gains[j]
// ---------------------------------------------------------------------------
__global__ __launch_bounds__(256) void apply_gains_kernel(
    _Float16* __restrict__ ctx, const float* __restrict__ gains)
{
    const int i = blockIdx.x * 256 + threadIdx.x;
    ctx[i] = (_Float16)((float)ctx[i] * gains[i & (HID_ - 1)]);
}

// ---------------------------------------------------------------------------
extern "C" void kernel_launch(void* const* d_in, const int* in_sizes, int n_in,
                              void* d_out, int out_size, void* d_ws, size_t ws_size,
                              hipStream_t stream)
{
    (void)in_sizes; (void)n_in; (void)out_size; (void)ws_size;
    const float* x    = (const float*)d_in[0];
    const float* W_in = (const float*)d_in[1];
    const float* b_in = (const float*)d_in[2];
    const float* Wg   = (const float*)d_in[3];
    const float* bg   = (const float*)d_in[4];
    const float* We   = (const float*)d_in[5];   // [16, 2176, 128]
    const float* be   = (const float*)d_in[6];   // flat [2048]
    const float* Wo   = (const float*)d_in[7];
    const float* bo   = (const float*)d_in[8];
    const float* Wh   = (const float*)d_in[9];   // [3, 2048, 2048]
    const float* bh   = (const float*)d_in[10];
    const float* Wcls = (const float*)d_in[11];
    const float* bcls = (const float*)d_in[12];
    float* out = (float*)d_out;

    // ---- workspace carve-up (halfs first, floats after) ----
    _Float16* h = (_Float16*)d_ws;
    size_t o = 0;
    _Float16* xh     = h + o; o += (size_t)B_ROWS * D_IN_;
    _Float16* WinT   = h + o; o += (size_t)HID_ * D_IN_;          // [2048][1024]
    _Float16* WeT    = h + o; o += (size_t)HID_ * ENH_;           // [2048][2176]
    _Float16* WoT    = h + o; o += (size_t)HID_ * EXP_DIM_;       // [2048][128]
    _Float16* WhT    = h + o; o += (size_t)3 * HID_ * HID_;       // 3x[2048][2048]
    _Float16* WclsT  = h + o; o += (size_t)N_CLS_ * HID_;         // [1000][2048]
    _Float16* enh    = h + o; o += (size_t)B_ROWS * ENH_;
    _Float16* expb   = h + o; o += (size_t)B_ROWS * HID_;         // expert / h1
    _Float16* comb   = h + o; o += (size_t)B_ROWS * EXP_DIM_;
    _Float16* ctx    = h + o; o += (size_t)B_ROWS * HID_;         // ctx / h3
    float* fp   = (float*)(h + o);
    float* gate = fp;                   // [B,16]
    float* gns  = fp + (size_t)B_ROWS * N_EXP_;                   // [2048]

    const dim3 blk(256);
    const int mT = B_ROWS / 128;        // 64

    // ---- one-shot f16 conversion / weight transposition ----
    convert_f16_kernel<<<(B_ROWS * D_IN_) / 256, blk, 0, stream>>>(x, xh);
    transpose_f16_kernel<<<(HID_ * D_IN_) / 256, blk, 0, stream>>>(W_in, WinT, D_IN_, HID_);
    for (int e = 0; e < N_EXP_; ++e)
        transpose_f16_kernel<<<(EXP_DIM_ * ENH_) / 256, blk, 0, stream>>>(
            We + (size_t)e * ENH_ * EXP_DIM_, WeT + (size_t)e * EXP_DIM_ * ENH_,
            ENH_, EXP_DIM_);
    transpose_f16_kernel<<<(HID_ * EXP_DIM_) / 256, blk, 0, stream>>>(Wo, WoT, EXP_DIM_, HID_);
    for (int i = 0; i < 3; ++i)
        transpose_f16_kernel<<<(HID_ * HID_) / 256, blk, 0, stream>>>(
            Wh + (size_t)i * HID_ * HID_, WhT + (size_t)i * HID_ * HID_, HID_, HID_);
    transpose_f16_kernel<<<(N_CLS_ * HID_) / 256, blk, 0, stream>>>(Wcls, WclsT, HID_, N_CLS_);

    // ---- 1. proj -> enh[:, :2048] ----
    wmma_gemm_f16<<<dim3(HID_ / 128, mT), blk, 0, stream>>>(
        xh, WinT, b_in, enh, B_ROWS, HID_, D_IN_, ENH_, 0, 1);
    // ---- 2. phasor bank ----
    phasor_kernel<<<B_ROWS, blk, 0, stream>>>(enh);
    // ---- 3. gate softmax ----
    gate_softmax_kernel<<<B_ROWS / 8, blk, 0, stream>>>(enh, Wg, bg, gate);
    // ---- 4. experts (plain transposed-B GEMM now) ----
    wmma_gemm_f16<<<dim3(HID_ / 128, mT), blk, 0, stream>>>(
        enh, WeT, be, expb, B_ROWS, HID_, ENH_, HID_, 1, 1);
    // ---- 5. gate-weighted combine ----
    moe_combine_kernel<<<(B_ROWS * EXP_DIM_) / 256, blk, 0, stream>>>(gate, expb, comb);
    // ---- 6. context = combined @ Wo + bo ----
    wmma_gemm_f16<<<dim3(HID_ / 128, mT), blk, 0, stream>>>(
        comb, WoT, bo, ctx, B_ROWS, HID_, EXP_DIM_, HID_, 0, 1);
    // ---- 7. spiking top-k -> gains ----
    spike_topk_kernel<<<1, blk, 0, stream>>>(ctx, gns);
    // ---- 8. attended = context * gains ----
    apply_gains_kernel<<<(B_ROWS * HID_) / 256, blk, 0, stream>>>(ctx, gns);
    // ---- 9. MLP x3 (ping-pong: ctx -> expb -> enh -> ctx) ----
    wmma_gemm_f16<<<dim3(HID_ / 128, mT), blk, 0, stream>>>(
        ctx, WhT, bh, expb, B_ROWS, HID_, HID_, HID_, 1, 1);
    wmma_gemm_f16<<<dim3(HID_ / 128, mT), blk, 0, stream>>>(
        expb, WhT + (size_t)HID_ * HID_, bh + HID_, enh, B_ROWS, HID_, HID_, HID_, 1, 1);
    wmma_gemm_f16<<<dim3(HID_ / 128, mT), blk, 0, stream>>>(
        enh, WhT + 2 * (size_t)HID_ * HID_, bh + 2 * HID_, ctx, B_ROWS, HID_, HID_, HID_, 1, 1);
    // ---- 10. classifier (N=1000, guarded tiles) -> f32 out ----
    wmma_gemm_f16<<<dim3((N_CLS_ + 127) / 128, mT), blk, 0, stream>>>(
        ctx, WclsT, bcls, out, B_ROWS, N_CLS_, HID_, N_CLS_, 0, 0);
}
